// RNN_1511828488779
// MI455X (gfx1250) — compile-verified, matrix operand order checked by reading, not run on hardware
//
#include <hip/hip_runtime.h>
#include <hip/hip_fp16.h>
#include <math.h>

typedef __attribute__((ext_vector_type(16))) _Float16 v16h;
typedef __attribute__((ext_vector_type(8)))  _Float16 v8h;
typedef __attribute__((ext_vector_type(4)))  _Float16 v4h;
typedef __attribute__((ext_vector_type(8)))  float    v8f;
typedef __attribute__((ext_vector_type(4)))  float    v4f;

#define B_   256
#define S_   512
#define DIN_ 64
#define H_   256

// ---- CDNA5 async global<->LDS copies (ASYNCcnt-tracked), guarded ----------
#if __has_builtin(__builtin_amdgcn_global_load_async_to_lds_b128) && \
    __has_builtin(__builtin_amdgcn_global_load_async_to_lds_b64)  && \
    __has_builtin(__builtin_amdgcn_global_store_async_from_lds_b128) && \
    __has_builtin(__builtin_amdgcn_s_wait_asynccnt)
#define USE_ASYNC 1
typedef int i32x2 __attribute__((ext_vector_type(2)));
typedef int i32x4 __attribute__((ext_vector_type(4)));
typedef __attribute__((address_space(1))) i32x2* gptr2_t;   // global
typedef __attribute__((address_space(3))) i32x2* lptr2_t;   // LDS
typedef __attribute__((address_space(1))) i32x4* gptr4_t;
typedef __attribute__((address_space(3))) i32x4* lptr4_t;
// Global side: flat-pointer integer value == global address -> inttoptr is exact.
// LDS side: C-style cast emits addrspacecast (strips aperture) -- required.
#define ASYNC_LD_B128(g, l)                                              \
    __builtin_amdgcn_global_load_async_to_lds_b128(                      \
        (gptr4_t)(unsigned long long)(g), (lptr4_t)(l), 0, 0)
#define ASYNC_LD_B64(g, l)                                               \
    __builtin_amdgcn_global_load_async_to_lds_b64(                       \
        (gptr2_t)(unsigned long long)(g), (lptr2_t)(l), 0, 0)
#define ASYNC_ST_B128(l, g)                                              \
    __builtin_amdgcn_global_store_async_from_lds_b128(                   \
        (gptr4_t)(unsigned long long)(g), (lptr4_t)(l), 0, 0)
#define ASYNC_WAIT() __builtin_amdgcn_s_wait_asynccnt(0)
#else
#define USE_ASYNC 0
#define ASYNC_WAIT()
#endif

// ---- hardware tanh (v_tanh_f32 on gfx1250), branchless fallback -----------
__device__ __forceinline__ float fast_tanh(float x) {
#if __has_builtin(__builtin_amdgcn_tanhf)
    return __builtin_amdgcn_tanhf(x);
#else
    float e = __expf(2.0f * x);
    return 1.0f - 2.0f / (e + 1.0f);
#endif
}

// Combine two 8-half LDS reads into one 16-half WMMA fragment.
__device__ __forceinline__ v16h load_frag(const _Float16* p0, const _Float16* p1) {
    v8h lo = *(const v8h*)p0;
    v8h hi = *(const v8h*)p1;
    return __builtin_shufflevector(lo, hi, 0,1,2,3,4,5,6,7,8,9,10,11,12,13,14,15);
}

// ---------------------------------------------------------------------------
// Pre-convert x [B,S,64] f32 -> f16 (bandwidth-bound, ~2us at 23.3 TB/s).
// Makes the layer-0 per-step prefetch a raw async byte copy.
// ---------------------------------------------------------------------------
__global__ __launch_bounds__(256) void x_to_f16(const float* __restrict__ x,
                                                _Float16* __restrict__ x16, int n4) {
    int i = blockIdx.x * 256 + threadIdx.x;
    if (i < n4) {
        v4f v = *(const v4f*)(x + (size_t)i * 4);
        v4h h;
        h.x = (_Float16)v.x; h.y = (_Float16)v.y;
        h.z = (_Float16)v.z; h.w = (_Float16)v.w;
        *(v4h*)(x16 + (size_t)i * 4) = h;
    }
}

// ---------------------------------------------------------------------------
// Layer 0: h_t = tanh(x_t @ W_ih^T + b + h_{t-1} @ W_hh^T), streams h out f16.
// 16 WGs x 256 threads (8 wave32); each WG owns 16 batch rows; K = 320 fused.
// ---------------------------------------------------------------------------
__global__ __launch_bounds__(256) void rnn_layer0(
    const _Float16* __restrict__ x16,   // [B,S,64] f16
    const float*    __restrict__ W_ih,  // [256,64]
    const float*    __restrict__ W_hh,  // [256,256]
    const float*    __restrict__ b_ih,
    const float*    __restrict__ b_hh,
    _Float16*       __restrict__ h_out) // [B,S,256] f16
{
    constexpr int K  = H_ + DIN_;   // 320
    constexpr int KS = K + 8;       // padded stride: b128 frag reads bank-conflict-free

    __shared__ _Float16 Wcat[H_][KS];       // [W_hh | W_ih] row-major f16 (~164 KB)
    __shared__ _Float16 Abuf[2][16][KS];    // [h_{t-1} | x_t], double-buffered
    __shared__ float    bias[H_];

    const int tid = threadIdx.x;
    const int b0  = blockIdx.x * 16;

    for (int idx = tid; idx < H_ * K; idx += 256) {
        int n = idx / K, k = idx % K;
        float v = (k < H_) ? W_hh[n * H_ + k] : W_ih[n * DIN_ + (k - H_)];
        Wcat[n][k] = (_Float16)v;
    }
    for (int n = tid; n < H_; n += 256) bias[n] = b_ih[n] + b_hh[n];

    // init buffer 0: h = 0, x-cols = x_0 (sync path, one time)
    for (int idx = tid; idx < 16 * H_; idx += 256)
        Abuf[0][idx >> 8][idx & 255] = (_Float16)0.f;
    {
        int row = tid >> 4, c4 = (tid & 15) * 4;
        *(v4h*)&Abuf[0][row][H_ + c4] =
            *(const v4h*)&x16[((size_t)(b0 + row) * S_) * DIN_ + c4];
    }
    __syncthreads();

    const int lane  = tid & 31;
    const int wave  = tid >> 5;
    const int mrow  = lane & 15;            // A-fragment row (M)
    const int khalf = (lane >> 4) << 3;     // lanes 16..31 hold the k+8 half
    const int nA    = wave * 32 + (lane & 15);   // tile-0 column
    const int nB    = nA + 16;                   // tile-1 column
    const int mbase = (lane >> 4) << 3;

    for (int t = 0; t < S_; ++t) {
        const int cur = t & 1, nxt = cur ^ 1;

        // prefetch x_{t+1} into nxt x-cols (8 B/lane, overlaps the WMMA work)
        if (t + 1 < S_) {
            int row = tid >> 4, c4 = (tid & 15) * 4;
            const _Float16* gp = &x16[((size_t)(b0 + row) * S_ + (t + 1)) * DIN_ + c4];
            _Float16*       lp = &Abuf[nxt][row][H_ + c4];
#if USE_ASYNC
            ASYNC_LD_B64(gp, lp);
#else
            *(v4h*)lp = *(const v4h*)gp;
#endif
        }

        // two 16x16 tiles per wave: shared A fragment, two independent chains
        float bb0 = bias[nA], bb1 = bias[nB];
        v8f acc0 = { bb0, bb0, bb0, bb0, bb0, bb0, bb0, bb0 };
        v8f acc1 = { bb1, bb1, bb1, bb1, bb1, bb1, bb1, bb1 };
        #pragma unroll
        for (int kc = 0; kc < K / 32; ++kc) {
            const int k0 = kc * 32 + khalf;
            v16h a   = load_frag(&Abuf[cur][mrow][k0], &Abuf[cur][mrow][k0 + 16]);
            v16h bf0 = load_frag(&Wcat[nA][k0], &Wcat[nA][k0 + 16]);
            v16h bf1 = load_frag(&Wcat[nB][k0], &Wcat[nB][k0 + 16]);
            acc0 = __builtin_amdgcn_wmma_f32_16x16x32_f16(
                       false, a, false, bf0, (short)0, acc0, false, false);
            acc1 = __builtin_amdgcn_wmma_f32_16x16x32_f16(
                       false, a, false, bf1, (short)0, acc1, false, false);
        }
        #pragma unroll
        for (int j = 0; j < 8; ++j) {
            int m = mbase + j;
            Abuf[nxt][m][nA] = (_Float16)fast_tanh(acc0[j]);
            Abuf[nxt][m][nB] = (_Float16)fast_tanh(acc1[j]);
        }

        ASYNC_WAIT();          // drain prefetch(t) + h-store(t-1) before handoff
        __syncthreads();

        // stream h_t out, coalesced, straight from LDS
        {
            int row = tid >> 4, c8 = (tid & 15) * 16;
            _Float16* lp = &Abuf[nxt][row][c8];
            _Float16* gp = &h_out[((size_t)(b0 + row) * S_ + t) * H_ + c8];
#if USE_ASYNC
            ASYNC_ST_B128(lp, gp);
            ASYNC_ST_B128(lp + 8, gp + 8);
#else
            *(v8h*)gp       = *(const v8h*)lp;
            *(v8h*)(gp + 8) = *(const v8h*)(lp + 8);
#endif
        }
    }
}

// ---------------------------------------------------------------------------
// Layer 1 + FC + sigmoid. Input is f16 h1 from layer 0. K = 512.
// ---------------------------------------------------------------------------
__global__ __launch_bounds__(256) void rnn_layer1_fc(
    const _Float16* __restrict__ h_in,   // [B,S,256] f16
    const float*    __restrict__ W_ih,   // [256,256]
    const float*    __restrict__ W_hh,   // [256,256]
    const float*    __restrict__ b_ih,
    const float*    __restrict__ b_hh,
    const float*    __restrict__ W_fc,   // [1,256]
    const float*    __restrict__ b_fc,   // [1]
    float*          __restrict__ out)    // [B] (B x 1)
{
    constexpr int K  = 2 * H_;      // 512
    constexpr int KS = K + 8;       // 520 padded stride

    __shared__ _Float16 Wcat[H_][KS];     // ~266 KB
    __shared__ _Float16 Abuf[2][16][KS];  // ~33 KB
    __shared__ float    bias[H_];

    const int tid = threadIdx.x;
    const int b0  = blockIdx.x * 16;

    for (int idx = tid; idx < H_ * K; idx += 256) {
        int n = idx >> 9, k = idx & (K - 1);
        float v = (k < H_) ? W_hh[n * H_ + k] : W_ih[n * H_ + (k - H_)];
        Wcat[n][k] = (_Float16)v;
    }
    for (int n = tid; n < H_; n += 256) bias[n] = b_ih[n] + b_hh[n];

    for (int idx = tid; idx < 16 * H_; idx += 256)
        Abuf[0][idx >> 8][idx & 255] = (_Float16)0.f;
    {
        int row = tid >> 4, c8 = (tid & 15) * 16;
        const _Float16* gp = &h_in[((size_t)(b0 + row) * S_) * H_ + c8];
        *(v8h*)&Abuf[0][row][H_ + c8]     = *(const v8h*)gp;
        *(v8h*)&Abuf[0][row][H_ + c8 + 8] = *(const v8h*)(gp + 8);
    }
    __syncthreads();

    const int lane  = tid & 31;
    const int wave  = tid >> 5;
    const int mrow  = lane & 15;
    const int khalf = (lane >> 4) << 3;
    const int nA    = wave * 32 + (lane & 15);
    const int nB    = nA + 16;
    const int mbase = (lane >> 4) << 3;

    for (int t = 0; t < S_; ++t) {
        const int cur = t & 1, nxt = cur ^ 1;

        if (t + 1 < S_) {   // prefetch h1_{t+1} (raw f16, 32 B/lane)
            int row = tid >> 4, c8 = (tid & 15) * 16;
            const _Float16* gp = &h_in[((size_t)(b0 + row) * S_ + (t + 1)) * H_ + c8];
            _Float16*       lp = &Abuf[nxt][row][H_ + c8];
#if USE_ASYNC
            ASYNC_LD_B128(gp, lp);
            ASYNC_LD_B128(gp + 8, lp + 8);
#else
            *(v8h*)lp       = *(const v8h*)gp;
            *(v8h*)(lp + 8) = *(const v8h*)(gp + 8);
#endif
        }

        float bb0 = bias[nA], bb1 = bias[nB];
        v8f acc0 = { bb0, bb0, bb0, bb0, bb0, bb0, bb0, bb0 };
        v8f acc1 = { bb1, bb1, bb1, bb1, bb1, bb1, bb1, bb1 };
        #pragma unroll
        for (int kc = 0; kc < K / 32; ++kc) {
            const int k0 = kc * 32 + khalf;
            v16h a   = load_frag(&Abuf[cur][mrow][k0], &Abuf[cur][mrow][k0 + 16]);
            v16h bf0 = load_frag(&Wcat[nA][k0], &Wcat[nA][k0 + 16]);
            v16h bf1 = load_frag(&Wcat[nB][k0], &Wcat[nB][k0 + 16]);
            acc0 = __builtin_amdgcn_wmma_f32_16x16x32_f16(
                       false, a, false, bf0, (short)0, acc0, false, false);
            acc1 = __builtin_amdgcn_wmma_f32_16x16x32_f16(
                       false, a, false, bf1, (short)0, acc1, false, false);
        }
        #pragma unroll
        for (int j = 0; j < 8; ++j) {
            int m = mbase + j;
            Abuf[nxt][m][nA] = (_Float16)fast_tanh(acc0[j]);
            Abuf[nxt][m][nB] = (_Float16)fast_tanh(acc1[j]);
        }

        ASYNC_WAIT();
        __syncthreads();
    }

    // S_ even -> final h lives in Abuf[0]. FC (D_OUT=1) + sigmoid.
    if (tid < 16) {
        float s = b_fc[0];
        for (int k = 0; k < H_; ++k)
            s += (float)Abuf[0][tid][k] * W_fc[k];
        out[b0 + tid] = 1.f / (1.f + __expf(-s));
    }
}

// ---------------------------------------------------------------------------
extern "C" void kernel_launch(void* const* d_in, const int* in_sizes, int n_in,
                              void* d_out, int out_size, void* d_ws, size_t ws_size,
                              hipStream_t stream) {
    const float* x     = (const float*)d_in[0];
    const float* W_ih0 = (const float*)d_in[1];
    const float* W_hh0 = (const float*)d_in[2];
    const float* b_ih0 = (const float*)d_in[3];
    const float* b_hh0 = (const float*)d_in[4];
    const float* W_ih1 = (const float*)d_in[5];
    const float* W_hh1 = (const float*)d_in[6];
    const float* b_ih1 = (const float*)d_in[7];
    const float* b_hh1 = (const float*)d_in[8];
    const float* W_fc  = (const float*)d_in[9];
    const float* b_fc  = (const float*)d_in[10];
    float* out = (float*)d_out;

    _Float16* x16 = (_Float16*)d_ws;                       // 16 MB
    _Float16* h1  = x16 + (size_t)B_ * S_ * DIN_;          // 64 MB

    const int n4 = B_ * S_ * DIN_ / 4;
    x_to_f16<<<n4 / 256, 256, 0, stream>>>(x, x16, n4);
    rnn_layer0<<<B_ / 16, 256, 0, stream>>>(x16, W_ih0, W_hh0, b_ih0, b_hh0, h1);
    rnn_layer1_fc<<<B_ / 16, 256, 0, stream>>>(h1, W_ih1, W_hh1, b_ih1, b_hh1,
                                               W_fc, b_fc, out);
}
